// Decoder_70643622085261
// MI455X (gfx1250) — matrix operand order repaired and checked by reading, not compiled, not run on hardware
//
#include <hip/hip_runtime.h>

// ---------------------------------------------------------------------------
// CDNA5 (gfx1250) wave32 WMMA implementation of a 3D transposed-conv decoder.
// Math: bf16 A/B, fp32 accumulate via v_wmma_f32_16x16x32_bf16.
// Each wave: M=16 output voxels x N=32 output channels (2 WMMA N-tiles),
// so one A fragment feeds two back-to-back WMMAs.
// ---------------------------------------------------------------------------

typedef __attribute__((ext_vector_type(16))) __bf16       v16bf;
typedef __attribute__((ext_vector_type(8)))  float        v8f;
typedef __attribute__((ext_vector_type(4)))  unsigned int u32x4;

union Frag { u32x4 q[2]; v16bf v; };
union Acc  { v8f v; float f[8]; };

static __device__ inline __bf16 f2bf(float f) {
  union { float f; unsigned u; } x; x.f = f;
  unsigned r = (x.u + 0x7FFFu + ((x.u >> 16) & 1u)) >> 16;  // RNE
  union { unsigned short s; __bf16 b; } y; y.s = (unsigned short)r;
  return y.b;
}

__global__ void f32_to_bf16_kernel(const float* __restrict__ src,
                                   __bf16* __restrict__ dst, int n) {
  int i = blockIdx.x * blockDim.x + threadIdx.x;
  if (i < n) dst[i] = f2bf(src[i]);
}

// src: fp32 [3][3][3][Cin][Cout] (DHWIO)  ->  dst: bf16 [27][CoutP][Cin]
__global__ void pack_weights_kernel(const float* __restrict__ src,
                                    __bf16* __restrict__ dst,
                                    int Cin, int Cout, int CoutP, int total) {
  int i = blockIdx.x * blockDim.x + threadIdx.x;
  if (i >= total) return;
  int cin  = i % Cin;
  int rest = i / Cin;
  int cp   = rest % CoutP;
  int tap  = rest / CoutP;
  float v = 0.f;
  if (cp < Cout) v = src[((size_t)tap * Cin + cin) * Cout + cp];
  dst[i] = f2bf(v);
}

// One wave per (16-output-voxel x 32-output-channel) tile.
// grid.x = Dout*Hout*(Wout/16), grid.y = CoutP/32, block = 32 (1 wave).
__global__ __launch_bounds__(32)
void deconv3d_wmma_kernel(const __bf16* __restrict__ in,
                          const __bf16* __restrict__ wpk,   // [27][CoutP][Cin] bf16
                          const float*  __restrict__ bias,  // [Cout] fp32
                          __bf16* __restrict__ out_bf,      // bf16 out (CoutP==Cout)
                          float*  __restrict__ out_f32,     // fp32 out (final layer)
                          int Din, int Hin, int Win,
                          int Dout, int Hout, int Wout,
                          int Cin, int Cout, int CoutP,
                          int stride, int relu) {
  const int lane = threadIdx.x;
  const int m    = lane & 15;   // M/N row within a 16x16 tile
  const int hi   = lane >> 4;   // K half selector

  const int wtiles = Wout >> 4;
  int t = blockIdx.x;
  const int x0 = (t % wtiles) << 4;
  t /= wtiles;
  const int y = t % Hout;
  const int z = t / Hout;
  const int n0 = (blockIdx.y << 5) + m;       // lane's channel, N-tile 0
  const int n1 = n0 + 16;                     // lane's channel, N-tile 1

  // Uniform taps in z and y.  conv_transpose (JAX SAME, k=3):
  //   stride 1: i = o + k - 1
  //   stride 2: i = (o + k - 2) / 2, valid iff (o+k) even
  int izs[3], kds[3]; int nkd = 0;
  int iys[3], khs[3]; int nkh = 0;
#pragma unroll
  for (int k = 0; k < 3; ++k) {
    if (stride == 2) {
      int tz = z + k - 2;
      if (!(tz & 1) && (unsigned)(tz >> 1) < (unsigned)Din) { kds[nkd] = k; izs[nkd] = tz >> 1; ++nkd; }
      int ty = y + k - 2;
      if (!(ty & 1) && (unsigned)(ty >> 1) < (unsigned)Hin) { khs[nkh] = k; iys[nkh] = ty >> 1; ++nkh; }
    } else {
      int tz = z + k - 1;
      if ((unsigned)tz < (unsigned)Din) { kds[nkd] = k; izs[nkd] = tz; ++nkd; }
      int ty = y + k - 1;
      if ((unsigned)ty < (unsigned)Hin) { khs[nkh] = k; iys[nkh] = ty; ++nkh; }
    }
  }
  // Per-lane taps in x (lane's M row = x0 + m).
  int ixs[3]; int vws[3];
#pragma unroll
  for (int k = 0; k < 3; ++k) {
    if (stride == 2) {
      int tx = (x0 + m) + k - 2;
      vws[k] = (!(tx & 1)) && ((unsigned)(tx >> 1) < (unsigned)Win);
      ixs[k] = tx >> 1;
    } else {
      int tx = (x0 + m) + k - 1;
      vws[k] = ((unsigned)tx < (unsigned)Win);
      ixs[k] = tx;
    }
  }

  Acc acc0{}, acc1{};  // fp32 accumulators for the two N-tiles

  for (int a = 0; a < nkd; ++a) {
    for (int b = 0; b < nkh; ++b) {
      const __bf16* in_row = in + ((size_t)izs[a] * Hin + iys[b]) * Win * Cin;
      const int tap_zy = (kds[a] * 3 + khs[b]) * 3;
#pragma unroll
      for (int kw = 0; kw < 3; ++kw) {
        const __bf16* wrow0 = wpk + ((size_t)(tap_zy + kw) * CoutP + n0) * Cin;
        const __bf16* wrow1 = wpk + ((size_t)(tap_zy + kw) * CoutP + n1) * Cin;
        const __bf16* arow  = in_row + (size_t)ixs[kw] * Cin;
        const int va = vws[kw];
        for (int c = 0; c < Cin; c += 32) {
          Frag A, B0, B1;
          // A 16x32 bf16: lane(hi=0): K = c+0..15 (elems 0..15)
          //               lane(hi=1): K = c+16..31
          if (va) {
            A.q[0] = *(const u32x4*)(arow + c + hi * 16);
            A.q[1] = *(const u32x4*)(arow + c + hi * 16 + 8);
          } else {
            u32x4 zz = {0u, 0u, 0u, 0u};
            A.q[0] = zz; A.q[1] = zz;
          }
          // B 32x16 bf16: lane N = n, elems 0..15 = K = c + hi*16 + 0..15
          const __bf16* bp0 = wrow0 + c + hi * 16;
          const __bf16* bp1 = wrow1 + c + hi * 16;
          B0.q[0] = *(const u32x4*)(bp0);
          B0.q[1] = *(const u32x4*)(bp0 + 8);
          B1.q[0] = *(const u32x4*)(bp1);
          B1.q[1] = *(const u32x4*)(bp1 + 8);
          acc0.v = __builtin_amdgcn_wmma_f32_16x16x32_bf16(
              false, A.v, false, B0.v, (short)0, acc0.v, false, false);
          acc1.v = __builtin_amdgcn_wmma_f32_16x16x32_bf16(
              false, A.v, false, B1.v, (short)0, acc1.v, false, false);
        }
      }
    }
  }

  // Epilogue: bias (+ReLU) and store.  D layout: lane N fixed; VGPR v -> M = v + 8*hi.
  const size_t row_base = ((size_t)z * Hout + y) * Wout + x0;
#pragma unroll
  for (int nt = 0; nt < 2; ++nt) {
    const int n = (nt == 0) ? n0 : n1;
    const Acc& acc = (nt == 0) ? acc0 : acc1;
    const float bv = (n < Cout) ? bias[n] : 0.f;
#pragma unroll
    for (int v = 0; v < 8; ++v) {
      const int mr = v + 8 * hi;
      float r = acc.f[v] + bv;
      if (relu) r = (r > 0.f) ? r : 0.f;
      const size_t idx = (row_base + mr) * (size_t)Cout + n;
      if (out_f32) {
        if (n < Cout) out_f32[idx] = r;
      } else {
        out_bf[idx] = f2bf(r);  // CoutP == Cout for all bf16 intermediate layers
      }
    }
  }
}

extern "C" void kernel_launch(void* const* d_in, const int* in_sizes, int n_in,
                              void* d_out, int out_size, void* d_ws, size_t ws_size,
                              hipStream_t stream) {
  (void)n_in; (void)out_size; (void)ws_size;
  const float* x   = (const float*)d_in[0];
  const float* w0  = (const float*)d_in[1];  const float* b0  = (const float*)d_in[2];
  const float* w00 = (const float*)d_in[3];  const float* b00 = (const float*)d_in[4];
  const float* w1  = (const float*)d_in[5];  const float* b1  = (const float*)d_in[6];
  const float* w10 = (const float*)d_in[7];  const float* b10 = (const float*)d_in[8];
  const float* w2  = (const float*)d_in[9];  const float* b2  = (const float*)d_in[10];
  const float* w20 = (const float*)d_in[11]; const float* b20 = (const float*)d_in[12];

  char* ws = (char*)d_ws;
  size_t off = 0;
  auto alloc = [&](size_t bytes) -> char* {
    char* p = ws + off;
    off += (bytes + 255) & ~(size_t)255;
    return p;
  };

  __bf16* xb   = (__bf16*)alloc((size_t)512 * 128 * 2);           // 8^3 x128
  __bf16* wp0  = (__bf16*)alloc((size_t)27 * 128 * 128 * 2);
  __bf16* wp00 = (__bf16*)alloc((size_t)27 * 128 * 128 * 2);
  __bf16* wp1  = (__bf16*)alloc((size_t)27 * 128 * 128 * 2);
  __bf16* wp10 = (__bf16*)alloc((size_t)27 * 128 * 128 * 2);
  __bf16* wp2  = (__bf16*)alloc((size_t)27 * 64  * 128 * 2);      // Cin128,CoutP64
  __bf16* wp20 = (__bf16*)alloc((size_t)27 * 32  * 64  * 2);      // Cin64, CoutP32
  __bf16* t0   = (__bf16*)alloc((size_t)16 * 16 * 16 * 128 * 2);  // L0 out
  __bf16* t1   = (__bf16*)alloc((size_t)16 * 16 * 16 * 128 * 2);  // L0_0 out
  __bf16* t2   = (__bf16*)alloc((size_t)32 * 32 * 32 * 128 * 2);  // L1 out
  __bf16* t3   = (__bf16*)alloc((size_t)32 * 32 * 32 * 128 * 2);  // L1_0 out
  __bf16* t4   = (__bf16*)alloc((size_t)64 * 64 * 64 * 64  * 2);  // L2 out

  // --- prep: x -> bf16, weights -> packed bf16 [tap][coutP][cin] ---
  {
    int n = in_sizes[0];  // 65536
    f32_to_bf16_kernel<<<(n + 255) / 256, 256, 0, stream>>>(x, xb, n);
  }
  auto pack = [&](const float* s, __bf16* d, int Cin, int Cout, int CoutP) {
    int total = 27 * CoutP * Cin;
    pack_weights_kernel<<<(total + 255) / 256, 256, 0, stream>>>(s, d, Cin, Cout, CoutP, total);
  };
  pack(w0,  wp0,  128, 128, 128);
  pack(w00, wp00, 128, 128, 128);
  pack(w1,  wp1,  128, 128, 128);
  pack(w10, wp10, 128, 128, 128);
  pack(w2,  wp2,  128, 64,  64);
  pack(w20, wp20, 64,  3,   32);

  auto layer = [&](const __bf16* in, const __bf16* w, const float* b,
                   __bf16* obf, float* of32,
                   int Din, int Hin, int Win, int Dout, int Hout, int Wout,
                   int Cin, int Cout, int CoutP, int stride, int relu) {
    dim3 grid((unsigned)(Dout * Hout * (Wout / 16)), (unsigned)(CoutP / 32), 1u);
    deconv3d_wmma_kernel<<<grid, 32, 0, stream>>>(
        in, w, b, obf, of32, Din, Hin, Win, Dout, Hout, Wout,
        Cin, Cout, CoutP, stride, relu);
  };

  // deconv0 (s2) -> deconv0_0 (s1) + ReLU
  layer(xb, wp0,  b0,  t0, nullptr,  8,  8,  8, 16, 16, 16, 128, 128, 128, 2, 0);
  layer(t0, wp00, b00, t1, nullptr, 16, 16, 16, 16, 16, 16, 128, 128, 128, 1, 1);
  // deconv1 (s2) -> deconv1_0 (s1) + ReLU
  layer(t1, wp1,  b1,  t2, nullptr, 16, 16, 16, 32, 32, 32, 128, 128, 128, 2, 0);
  layer(t2, wp10, b10, t3, nullptr, 32, 32, 32, 32, 32, 32, 128, 128, 128, 1, 1);
  // deconv2 (s2, 128->64) -> deconv2_0 (s1, 64->3), fp32 out
  layer(t3, wp2,  b2,  t4, nullptr, 32, 32, 32, 64, 64, 64, 128, 64, 64, 2, 0);
  layer(t4, wp20, b20, nullptr, (float*)d_out,
        64, 64, 64, 64, 64, 64, 64, 3, 32, 1, 0);
}
